// SelfAttention1d_22900765622729
// MI455X (gfx1250) — matrix operand order repaired and checked by reading, not compiled, not run on hardware
//
#include <hip/hip_runtime.h>

typedef __attribute__((ext_vector_type(16))) _Float16 v16h;
typedef __attribute__((ext_vector_type(8)))  float    v8f;

constexpr int cB = 4, cC = 512, cS = 2048, cH = 8, cD = 64;
constexpr int cM = cB * cS;   // 8192
constexpr int K_ = cC;        // 512

// ---------------- async global->LDS copy (gfx1250, ASYNCcnt) ----------------
// VDST vgpr = per-lane LDS byte offset (low 32 bits of shared-aperture addr),
// VADDR = per-lane 64-bit global address, 16 bytes per lane.
__device__ __forceinline__ void async_ld_b128(const void* gsrc, unsigned lds_off) {
  asm volatile("global_load_async_to_lds_b128 %0, %1, off"
               :
               : "v"(lds_off), "v"(gsrc)
               : "memory");
}
__device__ __forceinline__ void wait_async0() {
  asm volatile("s_wait_asynccnt 0x0" ::: "memory");
}
__device__ __forceinline__ unsigned lds_off(const void* p) {
  return (unsigned)(uintptr_t)p;  // low 32 bits of LDS generic address = LDS offset
}

// ---------------- WMMA fragment helpers (wave32, 16x16x32 f16) ----------------
// A-frag (16x32): lane m = lane&15; 16 contiguous K halfs; lanes>=16 -> K+16
// -> vectorizes to 2x ds_load_b128.
__device__ __forceinline__ v16h afrag_ld(const _Float16* rowp, int lane) {
  v16h r;
  const int hi = (lane & 16) ? 16 : 0;
#pragma unroll
  for (int e = 0; e < 16; ++e) {
    const int k = e + hi;
    r[e] = rowp[k];
  }
  return r;
}
// B-frag (32x16): lane n = lane&15; 16 contiguous K halfs; lanes>=16 -> K+16
__device__ __forceinline__ v16h bfrag_ld(const _Float16* p) {
  v16h r;
#pragma unroll
  for (int e = 0; e < 16; ++e) r[e] = p[e];
  return r;
}
__device__ __forceinline__ v8f wmma16(v16h a, v16h b, v8f c) {
  return __builtin_amdgcn_wmma_f32_16x16x32_f16(false, a, false, b, (short)0, c,
                                                false, false);
}
__device__ __forceinline__ v8f zero8() {
  v8f z = {0.f, 0.f, 0.f, 0.f, 0.f, 0.f, 0.f, 0.f};
  return z;
}

// ---------------- GroupNorm (1 group): per-batch mean/rstd ----------------
__global__ __launch_bounds__(256) void gn_stats_kernel(const float* __restrict__ x,
                                                       float* __restrict__ stats) {
  const int b = blockIdx.x;
  const int tid = threadIdx.x;
  const int N = cC * cS;
  const float* xb = x + (size_t)b * N;
  float s1 = 0.f, s2 = 0.f;
  for (int i = tid; i < N; i += 256) {
    float v = xb[i];
    s1 += v;
    s2 += v * v;
  }
  __shared__ float r1[256], r2[256];
  r1[tid] = s1;
  r2[tid] = s2;
  __syncthreads();
  for (int st = 128; st > 0; st >>= 1) {
    if (tid < st) { r1[tid] += r1[tid + st]; r2[tid] += r2[tid + st]; }
    __syncthreads();
  }
  if (tid == 0) {
    float mean = r1[0] / (float)N;
    float var = r2[0] / (float)N - mean * mean;
    stats[b * 2] = mean;
    stats[b * 2 + 1] = rsqrtf(var + 1e-5f);
  }
}

// normalize + affine + transpose [B,C,S] f32 -> [B,S,C] f16
__global__ __launch_bounds__(256) void gn_apply_kernel(const float* __restrict__ x,
                                                       const float* __restrict__ w,
                                                       const float* __restrict__ bias,
                                                       const float* __restrict__ stats,
                                                       _Float16* __restrict__ h) {
  const int idx = blockIdx.x * 256 + threadIdx.x;  // over B*C*S
  const int s = idx % cS;
  const int c = (idx / cS) % cC;
  const int b = idx / (cS * cC);
  const float mean = stats[b * 2], rstd = stats[b * 2 + 1];
  const float v = (x[idx] - mean) * rstd * w[c] + bias[c];
  h[((size_t)b * cS + s) * cC + c] = (_Float16)v;
}

__global__ __launch_bounds__(256) void f32_to_f16_kernel(const float* __restrict__ src,
                                                         _Float16* __restrict__ dst,
                                                         int n) {
  const int i = blockIdx.x * 256 + threadIdx.x;
  if (i < n) dst[i] = (_Float16)src[i];
}

// ---------------- Tiled WMMA GEMM: out = A[M,K] @ W[N,K]^T ----------------
// Double-buffered LDS, async global->LDS tile copies, prefetch two tiles ahead.
// MODE 0: Q (scale, head-split)  1: K (scale, head-split)  2: V (head-split)
// MODE 3: O-proj (+bias +residual, write f32 [B,C,S])
template <int MODE>
__global__ __launch_bounds__(256) void gemm_kernel(
    const _Float16* __restrict__ A, const _Float16* __restrict__ W,
    const float* __restrict__ bias, const float* __restrict__ resid,
    _Float16* __restrict__ outH, float* __restrict__ outF) {
  __shared__ __align__(16) _Float16 As[2][128 * 40];  // [m][k], stride 40 halfs
  __shared__ __align__(16) _Float16 Bs[2][128 * 40];  // [n][k], stride 40 halfs

  const int tid = threadIdx.x;
  const int lane = tid & 31;
  const int wave = tid >> 5;
  const int wm = wave >> 2;  // 0..1  (wave tile 64 x 32)
  const int wn = wave & 3;   // 0..3
  const int m0 = blockIdx.x * 128;
  const int n0 = blockIdx.y * 128;

  v8f acc[4][2];
#pragma unroll
  for (int i = 0; i < 4; ++i)
#pragma unroll
    for (int j = 0; j < 2; ++j) acc[i][j] = zero8();

  const int ldr = tid >> 1;        // 0..127 (tile row)
  const int ldc = (tid & 1) * 16;  // 0 / 16 (k half)
  const _Float16* agBase = A + (size_t)(m0 + ldr) * K_ + ldc;
  const _Float16* bgBase = W + (size_t)(n0 + ldr) * K_ + ldc;

  auto issue_tile = [&](int kt, int buf) {
    const _Float16* ag = agBase + kt * 32;
    const _Float16* bg = bgBase + kt * 32;
    const unsigned ao = lds_off(&As[buf][ldr * 40 + ldc]);
    const unsigned bo = lds_off(&Bs[buf][ldr * 40 + ldc]);
    async_ld_b128(ag, ao);
    async_ld_b128(ag + 8, ao + 16);
    async_ld_b128(bg, bo);
    async_ld_b128(bg + 8, bo + 16);
  };

  constexpr int NK = K_ / 32;
  issue_tile(0, 0);

  for (int kt = 0; kt < NK; ++kt) {
    const int cur = kt & 1;
    wait_async0();       // tile kt resident in LDS (this wave's copies)
    __syncthreads();     // ... and everyone else's; also: all reads of buf cur^1 done
    if (kt + 1 < NK) issue_tile(kt + 1, cur ^ 1);
    if (kt + 2 < NK) {   // warm L2 two tiles ahead
      __builtin_prefetch(agBase + (kt + 2) * 32, 0, 0);
      __builtin_prefetch(bgBase + (kt + 2) * 32, 0, 0);
    }

    v16h af[4], bf[2];
#pragma unroll
    for (int tm = 0; tm < 4; ++tm)
      af[tm] = afrag_ld(&As[cur][(wm * 64 + tm * 16 + (lane & 15)) * 40], lane);
#pragma unroll
    for (int tn = 0; tn < 2; ++tn)
      bf[tn] = bfrag_ld(&Bs[cur][(wn * 32 + tn * 16 + (lane & 15)) * 40 +
                                 ((lane & 16) ? 16 : 0)]);
#pragma unroll
    for (int tm = 0; tm < 4; ++tm)
#pragma unroll
      for (int tn = 0; tn < 2; ++tn) acc[tm][tn] = wmma16(af[tm], bf[tn], acc[tm][tn]);
  }

  const float scl = (MODE <= 1) ? 0.35355339059327373f : 1.0f;  // 64^-0.25
#pragma unroll
  for (int tm = 0; tm < 4; ++tm) {
#pragma unroll
    for (int tn = 0; tn < 2; ++tn) {
#pragma unroll
      for (int r = 0; r < 8; ++r) {
        const int mg = m0 + wm * 64 + tm * 16 + r + ((lane & 16) ? 8 : 0);
        const int ng = n0 + wn * 32 + tn * 16 + (lane & 15);
        const int b = mg >> 11;      // / S
        const int s = mg & (cS - 1);
        float v = acc[tm][tn][r] + bias[ng];
        if (MODE == 3) {
          const size_t oi = ((size_t)b * cC + ng) * cS + s;
          outF[oi] = v + resid[oi];
        } else {
          v *= scl;
          const int hh = ng >> 6, d = ng & 63;
          outH[(((size_t)b * cH + hh) * cS + s) * cD + d] = (_Float16)v;
        }
      }
    }
  }
}

// ---------------- Flash attention: per (b,h), streaming softmax ----------------
// block = 128 threads = 4 waves; each wave owns 16 q rows; block covers 64 rows.
// K chunk staged via async global->LDS; V staged transposed (manual, scatter).
__global__ __launch_bounds__(128) void attn_kernel(const _Float16* __restrict__ Q,
                                                   const _Float16* __restrict__ Kb,
                                                   const _Float16* __restrict__ Vb,
                                                   _Float16* __restrict__ AO) {
  __shared__ __align__(16) _Float16 Ks[32 * 72];       // [k][d] stride 72
  __shared__ __align__(16) _Float16 VsT[64 * 40];      // [d][k] stride 40
  __shared__ __align__(16) _Float16 Ps[4 * 16 * 40];   // per-wave P [16 x 32]

  const int tid = threadIdx.x;
  const int lane = tid & 31;
  const int wave = tid >> 5;  // 0..3
  const int bh = blockIdx.y;
  const int b = bh >> 3;   // / H
  const int hh = bh & 7;
  const int q0 = blockIdx.x * 64 + wave * 16;

  const _Float16* Qp = Q + (size_t)bh * cS * cD;
  const _Float16* Kp = Kb + (size_t)bh * cS * cD;
  const _Float16* Vp = Vb + (size_t)bh * cS * cD;

  // Q fragments held in registers for the whole pass (2 x 16x32 over d)
  v16h qf[2];
#pragma unroll
  for (int dc = 0; dc < 2; ++dc)
    qf[dc] = afrag_ld(Qp + (size_t)(q0 + (lane & 15)) * cD + dc * 32, lane);

  v8f accO[4];
#pragma unroll
  for (int i = 0; i < 4; ++i) accO[i] = zero8();
  float mrow[8], lrow[8];
#pragma unroll
  for (int r = 0; r < 8; ++r) { mrow[r] = -1e30f; lrow[r] = 0.f; }

  const int kk = tid & 31;          // loader: K/V row in chunk
  const int dseg = (tid >> 5) * 16; // loader: d segment
  const unsigned ko = lds_off(&Ks[kk * 72 + dseg]);

  for (int kc = 0; kc < cS / 32; ++kc) {
    // async K chunk [32 x 64] row-major
    const _Float16* kg = Kp + (size_t)(kc * 32 + kk) * cD + dseg;
    async_ld_b128(kg, ko);
    async_ld_b128(kg + 8, ko + 16);
    // V transposed staging [d][k] (2-byte scatter; async can't transpose)
    const _Float16* vg = Vp + (size_t)(kc * 32 + kk) * cD + dseg;
#pragma unroll
    for (int e = 0; e < 16; ++e) VsT[(dseg + e) * 40 + kk] = vg[e];
    if (kc + 1 < cS / 32) {  // warm L2 for next chunk
      __builtin_prefetch(kg + 32 * cD, 0, 0);
      __builtin_prefetch(vg + 32 * cD, 0, 0);
    }
    wait_async0();
    __syncthreads();

    // scores: S = Q @ K^T  (two 16x16 tiles over this 32-k chunk)
    v8f sf[2];
#pragma unroll
    for (int kt = 0; kt < 2; ++kt) {
      v8f s = zero8();
#pragma unroll
      for (int dc = 0; dc < 2; ++dc) {
        v16h bf = bfrag_ld(&Ks[(kt * 16 + (lane & 15)) * 72 + dc * 32 +
                               ((lane & 16) ? 16 : 0)]);
        s = wmma16(qf[dc], bf, s);
      }
      sf[kt] = s;
    }

    // online softmax: C-frag row r lives in lanes 0-15 (row r) / 16-31 (row r+8)
    float corr[8];
#pragma unroll
    for (int r = 0; r < 8; ++r) {
      float v0 = sf[0][r], v1 = sf[1][r];
      float mx = fmaxf(v0, v1);
#pragma unroll
      for (int msk = 1; msk < 16; msk <<= 1)
        mx = fmaxf(mx, __shfl_xor(mx, msk, 32));
      const float mn = fmaxf(mrow[r], mx);
      const float c = __expf(mrow[r] - mn);
      const float p0 = __expf(v0 - mn);
      const float p1 = __expf(v1 - mn);
      float rs = p0 + p1;
#pragma unroll
      for (int msk = 1; msk < 16; msk <<= 1) rs += __shfl_xor(rs, msk, 32);
      lrow[r] = lrow[r] * c + rs;
      mrow[r] = mn;
      corr[r] = c;
      sf[0][r] = p0;
      sf[1][r] = p1;
    }
#pragma unroll
    for (int dt = 0; dt < 4; ++dt)
#pragma unroll
      for (int r = 0; r < 8; ++r) accO[dt][r] *= corr[r];

    // re-shape P (C-frag) -> A-frag via per-wave LDS region (in-order DS ops)
    _Float16* pw = &Ps[wave * 16 * 40];
    const int prow = (lane & 16) ? 8 : 0;
#pragma unroll
    for (int kt = 0; kt < 2; ++kt)
#pragma unroll
      for (int r = 0; r < 8; ++r)
        pw[(r + prow) * 40 + kt * 16 + (lane & 15)] = (_Float16)sf[kt][r];

    v16h pf = afrag_ld(&pw[(lane & 15) * 40], lane);
#pragma unroll
    for (int dt = 0; dt < 4; ++dt) {
      v16h vf = bfrag_ld(&VsT[(dt * 16 + (lane & 15)) * 40 +
                              ((lane & 16) ? 16 : 0)]);
      accO[dt] = wmma16(pf, vf, accO[dt]);
    }
    __syncthreads();
  }

  // O / l  ->  AO[b, s, hh*64 + d] f16
#pragma unroll
  for (int dt = 0; dt < 4; ++dt) {
#pragma unroll
    for (int r = 0; r < 8; ++r) {
      const int s = q0 + r + ((lane & 16) ? 8 : 0);
      const int c = hh * 64 + dt * 16 + (lane & 15);
      AO[((size_t)b * cS + s) * cC + c] = (_Float16)(accO[dt][r] / lrow[r]);
    }
  }
}

// ---------------- launch ----------------
extern "C" void kernel_launch(void* const* d_in, const int* in_sizes, int n_in,
                              void* d_out, int out_size, void* d_ws, size_t ws_size,
                              hipStream_t stream) {
  (void)in_sizes; (void)n_in; (void)out_size; (void)ws_size;
  const float* hs  = (const float*)d_in[0];
  const float* gnw = (const float*)d_in[1];
  const float* gnb = (const float*)d_in[2];
  const float* wq  = (const float*)d_in[3];
  const float* bq  = (const float*)d_in[4];
  const float* wk  = (const float*)d_in[5];
  const float* bk  = (const float*)d_in[6];
  const float* wv  = (const float*)d_in[7];
  const float* bv  = (const float*)d_in[8];
  const float* wo  = (const float*)d_in[9];
  const float* bo  = (const float*)d_in[10];
  float* out = (float*)d_out;

  char* ws = (char*)d_ws;
  size_t off = 0;
  auto alloc = [&](size_t bytes) -> void* {
    void* p = ws + off;
    off += (bytes + 255) & ~(size_t)255;
    return p;
  };
  float*    stats = (float*)alloc(2 * cB * sizeof(float));
  _Float16* h16   = (_Float16*)alloc((size_t)cM * cC * 2);
  _Float16* wq16  = (_Float16*)alloc((size_t)cC * cC * 2);
  _Float16* wk16  = (_Float16*)alloc((size_t)cC * cC * 2);
  _Float16* wv16  = (_Float16*)alloc((size_t)cC * cC * 2);
  _Float16* wo16  = (_Float16*)alloc((size_t)cC * cC * 2);
  _Float16* Q16   = (_Float16*)alloc((size_t)cM * cC * 2);
  _Float16* K16   = (_Float16*)alloc((size_t)cM * cC * 2);
  _Float16* V16   = (_Float16*)alloc((size_t)cM * cC * 2);
  _Float16* AO16  = (_Float16*)alloc((size_t)cM * cC * 2);

  gn_stats_kernel<<<cB, 256, 0, stream>>>(hs, stats);
  gn_apply_kernel<<<(cB * cC * cS) / 256, 256, 0, stream>>>(hs, gnw, gnb, stats, h16);

  const int wn = cC * cC;
  f32_to_f16_kernel<<<(wn + 255) / 256, 256, 0, stream>>>(wq, wq16, wn);
  f32_to_f16_kernel<<<(wn + 255) / 256, 256, 0, stream>>>(wk, wk16, wn);
  f32_to_f16_kernel<<<(wn + 255) / 256, 256, 0, stream>>>(wv, wv16, wn);
  f32_to_f16_kernel<<<(wn + 255) / 256, 256, 0, stream>>>(wo, wo16, wn);

  dim3 gg(cM / 128, cC / 128);
  gemm_kernel<0><<<gg, 256, 0, stream>>>(h16, wq16, bq, nullptr, Q16, nullptr);
  gemm_kernel<1><<<gg, 256, 0, stream>>>(h16, wk16, bk, nullptr, K16, nullptr);
  gemm_kernel<2><<<gg, 256, 0, stream>>>(h16, wv16, bv, nullptr, V16, nullptr);

  attn_kernel<<<dim3(cS / 64, cB * cH), 128, 0, stream>>>(Q16, K16, V16, AO16);

  gemm_kernel<3><<<gg, 256, 0, stream>>>(AO16, wo16, bo, hs, nullptr, out);
}
